// ViT_UNet_57664230916962
// MI455X (gfx1250) — matrix operand order repaired and checked by reading, not compiled, not run on hardware
//
#include <hip/hip_runtime.h>
#include <hip/hip_bf16.h>

// ---------------------------------------------------------------------------
// DeepViT ReAttention transformer block for MI455X (gfx1250, wave32, WMMA).
// All large GEMMs run on v_wmma_f32_16x16x32_f16 (f16 in, f32 accumulate).
// 128x128 block tile, 8 waves x (32x64), double-buffered LDS, 1 barrier/step.
// Softmax / LayerNorm are single-pass, register-resident.
// ---------------------------------------------------------------------------

#define BB   4
#define NN   1024
#define CH   3
#define SS   16
#define HEADS 8
#define CC   768            // CH*SS*SS
#define HD   96             // CC / HEADS
#define HID  3072
#define BH   (BB*HEADS)     // 32
#define BN   (BB*NN)        // 4096
#define EPSF 1.0e-3f

typedef __attribute__((ext_vector_type(16))) _Float16 v16h;
typedef __attribute__((ext_vector_type(8)))  float    v8f;

struct U4 { unsigned int a, b, c, d; };           // POD 16-byte vector
struct U2 { unsigned int a, b; };                 // POD 8-byte vector
union F16Frag { v16h v; _Float16 h[16]; U4 q[2]; };
union H4 { U2 u; _Float16 h[4]; };

// tile config: 128x128 block tile, K step 32, 8 waves, each wave -> 32x64 slab
#define TM 128
#define TN 128
#define TK 32
#define LDT 40              // halves per LDS tile row (32 + 8 pad), 80B keeps 16B align
#define TILEH (TM * LDT)    // halves per (A or B) tile buffer

// Epilogues:
//  0: out_f16 = acc * scale
//  1: out_f16 = (acc + bias[n]) * (aux1[n]*rsqrt(1+eps)) + aux2[n]   (ReAttention)
//  3: out_f32 = acc + bias[n] + aux1[m*ldc+n]                        (bias+residual)
//  4: out_f16 = gelu_exact(acc + bias[n])                            (FFN1)
template<int EPI, bool BOUNDS>
__global__ __launch_bounds__(256)
void gemm_wmma_kernel(const _Float16* __restrict__ A,   // [M x K] row-major (+ z*sA)
                      const _Float16* __restrict__ Bt,  // [Nc x K] row-major, i.e. B^T (+ z*sB)
                      void* __restrict__ Out,
                      int M, int Nc, int K, int ldc,
                      long sA, long sB, int zdiv, long sO1, long sO2,
                      float scale,
                      const float* __restrict__ bias,
                      const float* __restrict__ aux1,
                      const float* __restrict__ aux2)
{
    const int z = blockIdx.z;
    A  += (long)z * sA;
    Bt += (long)z * sB;
    const long outOff = (long)(z / zdiv) * sO1 + (long)(z % zdiv) * sO2;

    const int tileM = blockIdx.y * TM;
    const int tileN = blockIdx.x * TN;

    __shared__ _Float16 lA[2 * TILEH];
    __shared__ _Float16 lB[2 * TILEH];

    const int t    = threadIdx.x;
    const int lane = t & 31;
    const int w    = t >> 5;
    const int wm   = (w >> 1) * 32;   // wave row strip within tile (0,32,64,96)
    const int wn   = (w & 1) * 64;    // wave col group within tile (0,64)

    // cooperative-loader indices: 256 threads cover 128 rows x 2 chunks of 16 halves
    const int lrow = t & 127;
    const int lko  = (t >> 7) * 16;   // 0 or 16

    v8f acc[2][4];
#pragma unroll
    for (int i = 0; i < 2; ++i)
#pragma unroll
        for (int j = 0; j < 4; ++j) acc[i][j] = (v8f){};

    const U4 z4 = {0u, 0u, 0u, 0u};
    U4 ra0, ra1, rb0, rb1;

    auto gload = [&](int k0) {
        if (!BOUNDS || tileM + lrow < M) {
            const _Float16* pa = A + (long)(tileM + lrow) * K + k0 + lko;
            ra0 = *(const U4*)pa;
            ra1 = *(const U4*)(pa + 8);
        } else { ra0 = z4; ra1 = z4; }
        if (!BOUNDS || tileN + lrow < Nc) {
            const _Float16* pb = Bt + (long)(tileN + lrow) * K + k0 + lko;
            rb0 = *(const U4*)pb;
            rb1 = *(const U4*)(pb + 8);
        } else { rb0 = z4; rb1 = z4; }
    };

    gload(0);
    int cur = 0;

    for (int k0 = 0; k0 < K; k0 += TK) {
        _Float16* bA = lA + cur * TILEH;
        _Float16* bB = lB + cur * TILEH;
        {   // stage registers -> LDS (current buffer)
            _Float16* pa = bA + lrow * LDT + lko;
            *(U4*)pa = ra0;
            *(U4*)(pa + 8) = ra1;
            _Float16* pb = bB + lrow * LDT + lko;
            *(U4*)pb = rb0;
            *(U4*)(pb + 8) = rb1;
        }
        __syncthreads();

        if (k0 + TK < K) gload(k0 + TK);          // overlap next global load
        if (!BOUNDS && k0 + 2 * TK < K) {         // gfx1250 global_prefetch_b8, 2 slabs ahead
            __builtin_prefetch(A  + (long)(tileM + lrow) * K + k0 + 2 * TK + lko, 0, 1);
            __builtin_prefetch(Bt + (long)(tileN + lrow) * K + k0 + 2 * TK + lko, 0, 1);
        }

        // --- fragments (per CDNA5 16-bit WMMA VGPR layouts) ---
        const int row  = lane & 15;
        const int kloA = (lane >> 4) * 8;    // A: halves 0..7 -> k=kloA.., 8..15 -> 16+kloA..
        const int kloB = (lane >> 4) * 8;    // B: halves 0..7 -> k=kloB.., 8..15 -> 16+kloB..

        F16Frag af[2], bf[4];
#pragma unroll
        for (int i = 0; i < 2; ++i) {
            const _Float16* p = bA + (wm + 16 * i + row) * LDT;
            af[i].q[0] = *(const U4*)(p + kloA);
            af[i].q[1] = *(const U4*)(p + 16 + kloA);
        }
#pragma unroll
        for (int j = 0; j < 4; ++j) {
            const _Float16* p = bB + (wn + 16 * j + row) * LDT;
            bf[j].q[0] = *(const U4*)(p + kloB);
            bf[j].q[1] = *(const U4*)(p + kloB + 16);
        }
#pragma unroll
        for (int i = 0; i < 2; ++i)
#pragma unroll
            for (int j = 0; j < 4; ++j)
                acc[i][j] = __builtin_amdgcn_wmma_f32_16x16x32_f16(
                    false, af[i].v, false, bf[j].v, (short)0, acc[i][j], false, false);

        cur ^= 1;
    }

    // --- epilogue; C/D layout: lane -> N=lane&15, VGPR r -> M = r + 8*(lane>>4)
    const int baseM = tileM + wm + 8 * (lane >> 4);
    const int baseN = tileN + wn + (lane & 15);

#pragma unroll
    for (int i = 0; i < 2; ++i) {
#pragma unroll
        for (int j = 0; j < 4; ++j) {
#pragma unroll
            for (int r = 0; r < 8; ++r) {
                const int gm = baseM + 16 * i + r;
                const int gn = baseN + 16 * j;
                if (BOUNDS && (gm >= M || gn >= Nc)) continue;
                const float v = acc[i][j][r];
                const long idx = outOff + (long)gm * ldc + gn;
                if constexpr (EPI == 0) {
                    ((_Float16*)Out)[idx] = (_Float16)(v * scale);
                } else if constexpr (EPI == 1) {
                    float y = (v + bias[gn]) * (aux1[gn] * rsqrtf(1.0f + EPSF)) + aux2[gn];
                    ((_Float16*)Out)[idx] = (_Float16)y;
                } else if constexpr (EPI == 3) {
                    ((float*)Out)[idx] = v + bias[gn] + aux1[(long)gm * ldc + gn];
                } else if constexpr (EPI == 4) {
                    float xg = v + bias[gn];
                    ((_Float16*)Out)[idx] =
                        (_Float16)(0.5f * xg * (1.0f + erff(xg * 0.70710678f)));
                }
            }
        }
    }
}

// ---------------------------------------------------------------------------
// per-patch 3x3 SAME conv producing q,k,v (fp16) + transposed v
// q,k,v: [B,H,N,HD]; vT: [B,H,HD,N]
// ---------------------------------------------------------------------------
__global__ __launch_bounds__(256)
void qkv_conv_kernel(const float* __restrict__ x,
                     const float* __restrict__ qw, const float* __restrict__ kw,
                     const float* __restrict__ vw,
                     _Float16* __restrict__ q, _Float16* __restrict__ k,
                     _Float16* __restrict__ v, _Float16* __restrict__ vT)
{
    const int patch = blockIdx.x;       // b*N + n
    const int b = patch >> 10;
    const int n = patch & (NN - 1);

    __shared__ float px[CC];            // 16x16x3 patch
    __shared__ float wq[81], wk[81], wv[81];

    const float* xp = x + (long)patch * CC;
    for (int i = threadIdx.x; i < CC; i += 256) px[i] = xp[i];
    if (threadIdx.x < 81) {
        wq[threadIdx.x] = qw[threadIdx.x];
        wk[threadIdx.x] = kw[threadIdx.x];
        wv[threadIdx.x] = vw[threadIdx.x];
    }
    __syncthreads();

    const int pix = threadIdx.x;        // one pixel per thread
    const int py = pix >> 4, pxx = pix & 15;

    float aq[CH] = {0.f, 0.f, 0.f}, ak[CH] = {0.f, 0.f, 0.f}, av[CH] = {0.f, 0.f, 0.f};
#pragma unroll
    for (int dy = -1; dy <= 1; ++dy) {
#pragma unroll
        for (int dx = -1; dx <= 1; ++dx) {
            const int yy = py + dy, xx = pxx + dx;
            if (yy < 0 || yy > 15 || xx < 0 || xx > 15) continue;   // SAME zero pad
            const float* pc = &px[(yy * SS + xx) * CH];
            const int wb = ((dy + 1) * 3 + (dx + 1)) * 9;           // [ky][kx][ci][co]
#pragma unroll
            for (int ci = 0; ci < CH; ++ci) {
                const float xv = pc[ci];
#pragma unroll
                for (int co = 0; co < CH; ++co) {
                    aq[co] += xv * wq[wb + ci * 3 + co];
                    ak[co] += xv * wk[wb + ci * 3 + co];
                    av[co] += xv * wv[wb + ci * 3 + co];
                }
            }
        }
    }
#pragma unroll
    for (int co = 0; co < CH; ++co) {
        const int g = pix * CH + co;            // global channel 0..767
        const int h = g / HD, d = g % HD;
        const long base = (((long)b * HEADS + h) * NN + n) * HD + d;
        q[base] = (_Float16)aq[co];
        k[base] = (_Float16)ak[co];
        v[base] = (_Float16)av[co];
        vT[(((long)b * HEADS + h) * HD + d) * NN + n] = (_Float16)av[co];
    }
}

// ---------------------------------------------------------------------------
// f32 [K x Nc] -> f16 transposed [Nc x K]
// ---------------------------------------------------------------------------
__global__ __launch_bounds__(256)
void convert_T_kernel(const float* __restrict__ W, _Float16* __restrict__ Wt,
                      int K, int Nc)
{
    const long i = (long)blockIdx.x * 256 + threadIdx.x;
    if (i < (long)K * Nc) {
        const int kk = (int)(i / Nc), nn = (int)(i % Nc);
        Wt[(long)nn * K + kk] = (_Float16)W[i];
    }
}

// ---------------------------------------------------------------------------
// single-pass in-place row softmax on fp16 probs (cols == 1024).
// Each thread owns 4 consecutive halves (one 8B load/store); reductions via
// wave32 shuffles + one LDS bounce. One block (256 threads) per row.
// ---------------------------------------------------------------------------
__global__ __launch_bounds__(256)
void softmax_rows_kernel(_Float16* __restrict__ P)
{
    const long row = blockIdx.x;
    _Float16* p = P + row * (long)NN;
    __shared__ float red[8];
    const int t = threadIdx.x, lane = t & 31, w = t >> 5;

    H4 hv;
    hv.u = *(const U2*)(p + t * 4);
    float f[4];
#pragma unroll
    for (int i = 0; i < 4; ++i) f[i] = (float)hv.h[i];

    float m = fmaxf(fmaxf(f[0], f[1]), fmaxf(f[2], f[3]));
#pragma unroll
    for (int off = 16; off > 0; off >>= 1) m = fmaxf(m, __shfl_xor(m, off, 32));
    if (lane == 0) red[w] = m;
    __syncthreads();
    m = red[0];
#pragma unroll
    for (int i = 1; i < 8; ++i) m = fmaxf(m, red[i]);
    __syncthreads();

    float e[4], s = 0.f;
#pragma unroll
    for (int i = 0; i < 4; ++i) { e[i] = __expf(f[i] - m); s += e[i]; }
#pragma unroll
    for (int off = 16; off > 0; off >>= 1) s += __shfl_xor(s, off, 32);
    if (lane == 0) red[w] = s;
    __syncthreads();
    s = 0.f;
#pragma unroll
    for (int i = 0; i < 8; ++i) s += red[i];
    const float inv = 1.0f / s;

#pragma unroll
    for (int i = 0; i < 4; ++i) hv.h[i] = (_Float16)(e[i] * inv);
    *(U2*)(p + t * 4) = hv.u;
}

// ---------------------------------------------------------------------------
// single-pass row LayerNorm (cols == 768): 3 register-cached floats/thread
// (j = t, t+256, t+512); writes f32 result and optional f16 copy.
// ---------------------------------------------------------------------------
__global__ __launch_bounds__(256)
void layernorm_kernel(const float* __restrict__ X, float* __restrict__ Yf,
                      _Float16* __restrict__ Yh,
                      const float* __restrict__ g, const float* __restrict__ be)
{
    const long row = blockIdx.x;
    const float* xr = X + row * (long)CC;
    __shared__ float red[8];
    const int t = threadIdx.x, lane = t & 31, w = t >> 5;

    float f[3];
#pragma unroll
    for (int i = 0; i < 3; ++i) f[i] = xr[t + 256 * i];

    float s = f[0] + f[1] + f[2];
#pragma unroll
    for (int off = 16; off > 0; off >>= 1) s += __shfl_xor(s, off, 32);
    if (lane == 0) red[w] = s;
    __syncthreads();
    s = 0.f;
#pragma unroll
    for (int i = 0; i < 8; ++i) s += red[i];
    const float mean = s * (1.0f / (float)CC);
    __syncthreads();

    float v = 0.f;
#pragma unroll
    for (int i = 0; i < 3; ++i) { const float d = f[i] - mean; v += d * d; }
#pragma unroll
    for (int off = 16; off > 0; off >>= 1) v += __shfl_xor(v, off, 32);
    if (lane == 0) red[w] = v;
    __syncthreads();
    v = 0.f;
#pragma unroll
    for (int i = 0; i < 8; ++i) v += red[i];
    const float inv = rsqrtf(v * (1.0f / (float)CC) + EPSF);

#pragma unroll
    for (int i = 0; i < 3; ++i) {
        const int j = t + 256 * i;
        const float y = (f[i] - mean) * inv * g[j] + be[j];
        if (Yf) Yf[row * (long)CC + j] = y;
        if (Yh) Yh[row * (long)CC + j] = (_Float16)y;
    }
}

// ---------------------------------------------------------------------------
extern "C" void kernel_launch(void* const* d_in, const int* in_sizes, int n_in,
                              void* d_out, int out_size, void* d_ws, size_t ws_size,
                              hipStream_t stream)
{
    (void)in_sizes; (void)n_in; (void)out_size; (void)ws_size;

    const float* x        = (const float*)d_in[0];
    const float* qw       = (const float*)d_in[1];
    const float* kw       = (const float*)d_in[2];
    const float* vw       = (const float*)d_in[3];
    const float* reattenW = (const float*)d_in[4];
    const float* reattenB = (const float*)d_in[5];
    const float* bn_gamma = (const float*)d_in[6];
    const float* bn_beta  = (const float*)d_in[7];
    const float* proj_W   = (const float*)d_in[8];
    const float* proj_b   = (const float*)d_in[9];
    const float* ln1_g    = (const float*)d_in[10];
    const float* ln1_b    = (const float*)d_in[11];
    const float* W1       = (const float*)d_in[12];
    const float* b1       = (const float*)d_in[13];
    const float* W2       = (const float*)d_in[14];
    const float* b2       = (const float*)d_in[15];
    const float* ln2_g    = (const float*)d_in[16];
    const float* ln2_b    = (const float*)d_in[17];

    // ---- workspace carve ----
    char* wp = (char*)d_ws;
    auto alloc = [&](size_t bytes) -> void* {
        void* p = (void*)wp;
        wp += (bytes + 255) & ~(size_t)255;
        return p;
    };
    _Float16* qh   = (_Float16*)alloc((size_t)BH * NN * HD * 2);
    _Float16* kh   = (_Float16*)alloc((size_t)BH * NN * HD * 2);
    _Float16* vh   = (_Float16*)alloc((size_t)BH * NN * HD * 2);
    _Float16* vT   = (_Float16*)alloc((size_t)BH * HD * NN * 2);
    _Float16* Sp   = (_Float16*)alloc((size_t)BH * NN * NN * 2);   // logits -> probs (in place)
    _Float16* A2   = (_Float16*)alloc((size_t)BH * NN * NN * 2);   // reattended attn
    _Float16* o16  = (_Float16*)alloc((size_t)BN * CC * 2);
    float*    yf   = (float*)   alloc((size_t)BN * CC * 4);        // proj + residual
    float*    h1f  = (float*)   alloc((size_t)BN * CC * 4);
    _Float16* h1h  = (_Float16*)alloc((size_t)BN * CC * 2);
    _Float16* ff16 = (_Float16*)alloc((size_t)BN * HID * 2);
    float*    zf   = (float*)   alloc((size_t)BN * CC * 4);        // ffn2 + residual
    _Float16* WrT  = (_Float16*)alloc((size_t)NN * NN * 2);
    _Float16* pWT  = (_Float16*)alloc((size_t)CC * CC * 2);
    _Float16* W1T  = (_Float16*)alloc((size_t)CC * HID * 2);
    _Float16* W2T  = (_Float16*)alloc((size_t)HID * CC * 2);

    // ---- weight conversion (f32 -> f16, transposed to [N x K]) ----
    convert_T_kernel<<<(NN * NN + 255) / 256, 256, 0, stream>>>(reattenW, WrT, NN, NN);
    convert_T_kernel<<<(CC * CC + 255) / 256, 256, 0, stream>>>(proj_W, pWT, CC, CC);
    convert_T_kernel<<<(CC * HID + 255) / 256, 256, 0, stream>>>(W1, W1T, CC, HID);
    convert_T_kernel<<<(HID * CC + 255) / 256, 256, 0, stream>>>(W2, W2T, HID, CC);

    // ---- q,k,v via per-patch 3x3 conv ----
    qkv_conv_kernel<<<BN, 256, 0, stream>>>(x, qw, kw, vw, qh, kh, vh, vT);

    const float scale = rsqrtf((float)HD);   // (C/H)^-0.5

    // ---- logits: S = Q K^T * scale  (per b,h)  [1024x1024], K=96 ----
    gemm_wmma_kernel<0, false><<<dim3(NN / TN, NN / TM, BH), 256, 0, stream>>>(
        qh, kh, (void*)Sp, NN, NN, HD, NN,
        (long)NN * HD, (long)NN * HD, 1, (long)NN * NN, 0,
        scale, nullptr, nullptr, nullptr);

    // ---- softmax over last axis, in place ----
    softmax_rows_kernel<<<BH * NN, 256, 0, stream>>>(Sp);

    // ---- ReAttention: A2 = (P @ Wr + b) * gamma/sqrt(1+eps) + beta ----
    gemm_wmma_kernel<1, false><<<dim3(NN / TN, NN / TM, BH), 256, 0, stream>>>(
        Sp, WrT, (void*)A2, NN, NN, NN, NN,
        (long)NN * NN, 0, 1, (long)NN * NN, 0,
        1.0f, reattenB, bn_gamma, bn_beta);

    // ---- O = A2 @ V  -> o16 [B,N,C] (head-interleaved store) ----
    gemm_wmma_kernel<0, true><<<dim3(1, NN / TM, BH), 256, 0, stream>>>(
        A2, vT, (void*)o16, NN, HD, NN, CC,
        (long)NN * NN, (long)HD * NN, HEADS, (long)NN * CC, (long)HD,
        1.0f, nullptr, nullptr, nullptr);

    // ---- proj: y = o @ Wp + b + x  (f32 out) ----
    gemm_wmma_kernel<3, false><<<dim3(CC / TN, BN / TM, 1), 256, 0, stream>>>(
        o16, pWT, (void*)yf, BN, CC, CC, CC,
        0, 0, 1, 0, 0,
        1.0f, proj_b, x, nullptr);

    // ---- h1 = LN(y) ----
    layernorm_kernel<<<BN, 256, 0, stream>>>(yf, h1f, h1h, ln1_g, ln1_b);

    // ---- FFN1: ff = gelu(h1 @ W1 + b1) ----
    gemm_wmma_kernel<4, false><<<dim3(HID / TN, BN / TM, 1), 256, 0, stream>>>(
        h1h, W1T, (void*)ff16, BN, HID, CC, HID,
        0, 0, 1, 0, 0,
        1.0f, b1, nullptr, nullptr);

    // ---- FFN2: z = ff @ W2 + b2 + h1 ----
    gemm_wmma_kernel<3, false><<<dim3(CC / TN, BN / TM, 1), 256, 0, stream>>>(
        ff16, W2T, (void*)zf, BN, CC, HID, CC,
        0, 0, 1, 0, 0,
        1.0f, b2, h1f, nullptr);

    // ---- out = LN(z) ----
    layernorm_kernel<<<BN, 256, 0, stream>>>(zf, (float*)d_out, nullptr, ln2_g, ln2_b);
}
